// SAGE2HopEncoder_34419867910898
// MI455X (gfx1250) — matrix operand order repaired and checked by reading, not compiled, not run on hardware
//
#include <hip/hip_runtime.h>
#include <hip/hip_bf16.h>

// ---------------------------------------------------------------------------
// SAGE 2-hop encoder for MI455X (gfx1250, wave32, WMMA).
//  - scatter/aggregate: hardware global_atomic_add_f32 (L2-resident dst)
//  - weights pre-packed once to bf16 in WMMA B-fragment layout
//  - double-buffered LDS staging: global load issued at step start (registers),
//    ds_store deferred to step end so the fetch overlaps the 4 WMMAs
//  - all GEMMs: v_wmma_f32_16x16x32_bf16 (bf16 A/B, fp32 accumulate)
//  - mean-scaling, BN affine + ReLU folded into the WMMA A-fragment load
// ---------------------------------------------------------------------------

typedef __attribute__((ext_vector_type(16))) __bf16 v16bf;
typedef __attribute__((ext_vector_type(8)))  float  v8f;

#define NNODES 100000
#define NEDGES 1600000
#define INCH   128
#define HDIM   256
#define EMBD   128

// ---------------------------------------------------------------- utilities

__global__ __launch_bounds__(256) void zero_f4(float4* __restrict__ p, long n4) {
    long i = (long)blockIdx.x * blockDim.x + threadIdx.x;
    if (i < n4) p[i] = make_float4(0.f, 0.f, 0.f, 0.f);
}

__global__ __launch_bounds__(256) void count_deg(const int* __restrict__ dst,
                                                 float* __restrict__ deg, int ne) {
    int e = blockIdx.x * blockDim.x + threadIdx.x;
    if (e < ne) unsafeAtomicAdd(&deg[dst[e]], 1.0f);
}

__global__ __launch_bounds__(256) void finish_deg(float* __restrict__ deg, int n) {
    int i = blockIdx.x * blockDim.x + threadIdx.x;
    if (i < n) deg[i] = 1.0f / fmaxf(deg[i], 1.0f);
}

// One block handles 8 edges; one lane per feature channel; hw fp32 atomics.
template<int F>
__global__ __launch_bounds__(F) void scatter_add(const float* __restrict__ x,
                                                 const int* __restrict__ src,
                                                 const int* __restrict__ dst,
                                                 float* __restrict__ agg, int ne) {
    int e0 = blockIdx.x * 8;
    int c  = threadIdx.x;
#pragma unroll
    for (int i = 0; i < 8; ++i) {
        int e = e0 + i;
        if (e >= ne) return;
        int s = src[e], d = dst[e];
        unsafeAtomicAdd(&agg[(size_t)d * F + c], x[(size_t)s * F + c]);
    }
}

// Pack W[K x H] (f32, row-major) into bf16 WMMA B-fragment layout:
// tile (kt, ct) = 32x16 block; 512 bf16 per tile, lane-major:
//   Wp[tile*512 + lane*16 + j] = W[(kt*32 + (lane>>4)*16 + j) * H + ct*16 + (lane&15)]
__global__ __launch_bounds__(256) void pack_w(const float* __restrict__ W,
                                              __bf16* __restrict__ Wp, int K, int H) {
    int g = blockIdx.x * 256 + threadIdx.x;
    if (g >= K * H) return;
    int tile = g >> 9;
    int r    = g & 511;
    int l    = r >> 4;
    int j    = r & 15;
    int CT   = H >> 4;
    int kt   = tile / CT;
    int ct   = tile - kt * CT;
    int k    = kt * 32 + (l >> 4) * 16 + j;
    int col  = ct * 16 + (l & 15);
    Wp[g] = (__bf16)W[(size_t)k * H + col];
}

// Per-column batchnorm stats (training-mode), folded into scale/shift:
//   y = x*scale[c] + shift[c];  scale = gamma*rsqrt(var+eps), shift = beta-mu*scale
__global__ __launch_bounds__(256) void bn_stats(const float* __restrict__ h,
                                                const float* __restrict__ gamma,
                                                const float* __restrict__ beta,
                                                float* __restrict__ scale,
                                                float* __restrict__ shift,
                                                int n, int Hd) {
    int c = blockIdx.x;
    float s = 0.f, q = 0.f;
    for (int i = threadIdx.x; i < n; i += 256) {
        float v = h[(size_t)i * Hd + c];
        s += v; q += v * v;
    }
    __shared__ float ss[256], qq[256];
    ss[threadIdx.x] = s; qq[threadIdx.x] = q;
    __syncthreads();
    for (int o = 128; o > 0; o >>= 1) {
        if (threadIdx.x < o) { ss[threadIdx.x] += ss[threadIdx.x + o];
                               qq[threadIdx.x] += qq[threadIdx.x + o]; }
        __syncthreads();
    }
    if (threadIdx.x == 0) {
        float mu  = ss[0] / (float)n;
        float var = qq[0] / (float)n - mu * mu;
        float r   = rsqrtf(var + 1e-5f);
        float sc  = r * gamma[c];
        scale[c] = sc;
        shift[c] = beta[c] - mu * sc;
    }
}

// ------------------------------------------------------------- WMMA GEMM
// Out[N x H] = act( A0[N x K0] (*rowscale / *affine+relu) @ W0[K0 x H]
//                 (+ A1[N x K1] @ W1[K1 x H]) + bias )
// 8 waves/block; each wave: 16 rows x 64 cols (4 x 16x16 accumulators).
// Packed-B chunks double-buffered through LDS; one barrier per 32-K step;
// next chunk's global fetch is in flight during this step's WMMAs.

template<bool DUAL, bool ROWSCALE, bool AFFINE, bool RELU_OUT>
__global__ __launch_bounds__(256) void gemm_wmma(
    const float* __restrict__ A0, const __bf16* __restrict__ W0p, int K0,
    const float* __restrict__ A1, const __bf16* __restrict__ W1p, int K1,
    const float* __restrict__ rowscale,
    const float* __restrict__ kscale, const float* __restrict__ kshift,
    const float* __restrict__ bias,
    float* __restrict__ Out, int N, int H)
{
    const int lane = threadIdx.x & 31;
    const int wave = threadIdx.x >> 5;
    const int row0 = blockIdx.x * 128 + wave * 16;
    const bool active = (row0 < N);         // wave-uniform (N % 16 == 0)
    const int col0 = blockIdx.y * 64;
    const int CT   = H >> 4;                // column tiles in packed W
    const int ct0  = col0 >> 4;
    const int m    = lane & 15;             // row within A tile / col within B,C
    const int hk   = lane >> 4;             // half-K selector

    __shared__ __align__(32) __bf16 sB[2][2048];  // double-buffered 4 KB chunks

    v8f acc[4] = {};

    float rs = 1.0f;
    if (ROWSCALE && active) rs = rowscale[row0 + m];

    const int steps0 = K0 >> 5;
    const int steps  = DUAL ? (steps0 + (K1 >> 5)) : steps0;

    auto chunk_ptr = [&](int s) -> const __bf16* {
        if (!DUAL || s < steps0)
            return W0p + ((size_t)(s * CT) + ct0) * 512;
        return W1p + ((size_t)((s - steps0) * CT) + ct0) * 512;
    };

    // preload first chunk straight into LDS
    *(uint4*)&sB[0][threadIdx.x * 8] = *(const uint4*)&chunk_ptr(0)[threadIdx.x * 8];
    __syncthreads();

    for (int s = 0; s < steps; ++s) {
        const int cur = s & 1;

        // issue next chunk's global fetch NOW (registers only; the wait for it
        // lands at the ds_store after the WMMAs below -> true overlap)
        uint4 pre;
        const bool havePre = (s + 1 < steps);
        if (havePre)
            pre = *(const uint4*)&chunk_ptr(s + 1)[threadIdx.x * 8];

        if (active) {
            const int  pass = (DUAL && s >= steps0) ? 1 : 0;
            const int  kb   = (pass ? (s - steps0) : s) * 32;
            const float* arow = (pass ? A1 : A0) +
                                (size_t)(row0 + m) * (pass ? K1 : K0);

            // ---- A fragment: 16-bit 16x32 layout.
            const int ka = kb + hk * 16;
            const int kc = kb + hk * 16 + 8;
            float av[16];
            *(float4*)&av[0]  = *(const float4*)(arow + ka);
            *(float4*)&av[4]  = *(const float4*)(arow + ka + 4);
            *(float4*)&av[8]  = *(const float4*)(arow + kc);
            *(float4*)&av[12] = *(const float4*)(arow + kc + 4);

            // prefetch next K-step of this row (never crosses the row end)
            if (s + 1 < steps && !(DUAL && s + 1 == steps0)) {
                __builtin_prefetch(arow + ka + 32, 0, 0);
                __builtin_prefetch(arow + kc + 32, 0, 0);
            }

            if (AFFINE && pass == 0) {       // BN affine + ReLU on A elements
                float sv[16], tv[16];
                *(float4*)&sv[0]  = *(const float4*)(kscale + ka);
                *(float4*)&sv[4]  = *(const float4*)(kscale + ka + 4);
                *(float4*)&sv[8]  = *(const float4*)(kscale + kc);
                *(float4*)&sv[12] = *(const float4*)(kscale + kc + 4);
                *(float4*)&tv[0]  = *(const float4*)(kshift + ka);
                *(float4*)&tv[4]  = *(const float4*)(kshift + ka + 4);
                *(float4*)&tv[8]  = *(const float4*)(kshift + kc);
                *(float4*)&tv[12] = *(const float4*)(kshift + kc + 4);
#pragma unroll
                for (int i = 0; i < 16; ++i)
                    av[i] = fmaxf(fmaf(av[i], sv[i], tv[i]), 0.f);
            }
            if (ROWSCALE && pass == 0) {
#pragma unroll
                for (int i = 0; i < 16; ++i) av[i] *= rs;
            }
            v16bf afrag;
#pragma unroll
            for (int i = 0; i < 16; ++i) afrag[i] = (__bf16)av[i];

            // ---- fetch all 4 B fragments first, then issue 4 WMMAs so the
            // ds_load latency overlaps matrix issue instead of serializing.
            v16bf bfrag[4];
#pragma unroll
            for (int t = 0; t < 4; ++t)
                bfrag[t] = *(const v16bf*)&sB[cur][t * 512 + lane * 16];
#pragma unroll
            for (int t = 0; t < 4; ++t)
                acc[t] = __builtin_amdgcn_wmma_f32_16x16x32_bf16(
                    /*neg_a=*/false, afrag, /*neg_b=*/false, bfrag[t],
                    /*c_mod=*/(short)0, acc[t],
                    /*reuse_a=*/false, /*reuse_b=*/false);
        }

        // commit the prefetched chunk into the other LDS buffer (no conflict
        // with this step's reads), then one barrier publishes it.
        if (havePre)
            *(uint4*)&sB[cur ^ 1][threadIdx.x * 8] = pre;
        __syncthreads();
    }

    if (!active) return;

    // ---- epilogue: C/D layout (lane = column, VGPR r = row r + 8*hk)
#pragma unroll
    for (int t = 0; t < 4; ++t) {
        const int col = col0 + t * 16 + m;
        const float bb = bias[col];
        float* op = Out + (size_t)row0 * H + col;
#pragma unroll
        for (int r = 0; r < 8; ++r) {
            const int mr = r + 8 * hk;
            float v = acc[t][r] + bb;
            if (RELU_OUT) v = fmaxf(v, 0.f);
            op[(size_t)mr * H] = v;
        }
    }
}

// ------------------------------------------------------------- launcher

extern "C" void kernel_launch(void* const* d_in, const int* in_sizes, int n_in,
                              void* d_out, int out_size, void* d_ws, size_t ws_size,
                              hipStream_t stream) {
    const float* x    = (const float*)d_in[0];
    const int*   ei   = (const int*)d_in[1];
    const int*   src  = ei;
    const int*   dst  = ei + NEDGES;
    const float* W1l  = (const float*)d_in[2];
    const float* W1r  = (const float*)d_in[3];
    const float* b1   = (const float*)d_in[4];
    const float* W2l  = (const float*)d_in[5];
    const float* W2r  = (const float*)d_in[6];
    const float* b2   = (const float*)d_in[7];
    const float* Wm1  = (const float*)d_in[8];
    const float* bm1  = (const float*)d_in[9];
    const float* gam  = (const float*)d_in[10];
    const float* bet  = (const float*)d_in[11];
    const float* Wm2  = (const float*)d_in[12];
    const float* bm2  = (const float*)d_in[13];
    float* out = (float*)d_out;

    // workspace layout (float units; all offsets 256-float aligned)
    float* ws = (float*)d_ws;
    size_t off = 0;
    float* deg     = ws + off; off += 100352;                 // N rounded up
    float* bnscale = ws + off; off += 256;
    float* bnshift = ws + off; off += 256;
    __bf16* pw     = (__bf16*)(ws + off); off += 147456;      // 294912 bf16
    float* buf0    = ws + off; off += (size_t)NNODES * HDIM;  // agg1 -> h2
    float* buf1    = ws + off; off += (size_t)NNODES * HDIM;  // h1
    float* buf2    = ws + off; off += (size_t)NNODES * HDIM;  // agg2 -> h3

    __bf16* pW1l = pw;            // 128x256 -> 32768
    __bf16* pW1r = pw + 32768;    // 32768
    __bf16* pW2l = pw + 65536;    // 256x256 -> 65536
    __bf16* pW2r = pw + 131072;   // 65536
    __bf16* pWm1 = pw + 196608;   // 65536
    __bf16* pWm2 = pw + 262144;   // 256x128 -> 32768

    const int gemmRows = (NNODES + 127) / 128;                // 782

    // 0) pack weights to bf16 B-fragment layout (tiny, once per launch)
    pack_w<<<(INCH * HDIM + 255) / 256, 256, 0, stream>>>(W1l, pW1l, INCH, HDIM);
    pack_w<<<(INCH * HDIM + 255) / 256, 256, 0, stream>>>(W1r, pW1r, INCH, HDIM);
    pack_w<<<(HDIM * HDIM + 255) / 256, 256, 0, stream>>>(W2l, pW2l, HDIM, HDIM);
    pack_w<<<(HDIM * HDIM + 255) / 256, 256, 0, stream>>>(W2r, pW2r, HDIM, HDIM);
    pack_w<<<(HDIM * HDIM + 255) / 256, 256, 0, stream>>>(Wm1, pWm1, HDIM, HDIM);
    pack_w<<<(HDIM * EMBD + 255) / 256, 256, 0, stream>>>(Wm2, pWm2, HDIM, EMBD);

    // 1) degrees
    zero_f4<<<(NNODES / 4 + 255) / 256, 256, 0, stream>>>((float4*)deg, NNODES / 4);
    count_deg<<<(NEDGES + 255) / 256, 256, 0, stream>>>(dst, deg, NEDGES);
    finish_deg<<<(NNODES + 255) / 256, 256, 0, stream>>>(deg, NNODES);

    // 2) layer-1 aggregate: agg1 = scatter_sum(x[src] -> dst)   [N x 128]
    {
        long n4 = (long)NNODES * INCH / 4;
        zero_f4<<<(unsigned)((n4 + 255) / 256), 256, 0, stream>>>((float4*)buf0, n4);
        scatter_add<INCH><<<(NEDGES + 7) / 8, INCH, 0, stream>>>(x, src, dst, buf0, NEDGES);
    }

    // 3) h1 = relu( (agg1*deginv) @ W1l + x @ W1r + b1 )        [N x 256]
    gemm_wmma<true, true, false, true><<<dim3(gemmRows, HDIM / 64), 256, 0, stream>>>(
        buf0, pW1l, INCH, x, pW1r, INCH, deg, nullptr, nullptr, b1, buf1, NNODES, HDIM);

    // 4) layer-2 aggregate: agg2 = scatter_sum(h1[src] -> dst)  [N x 256]
    {
        long n4 = (long)NNODES * HDIM / 4;
        zero_f4<<<(unsigned)((n4 + 255) / 256), 256, 0, stream>>>((float4*)buf2, n4);
        scatter_add<HDIM><<<(NEDGES + 7) / 8, HDIM, 0, stream>>>(buf1, src, dst, buf2, NEDGES);
    }

    // 5) h2 = relu( (agg2*deginv) @ W2l + h1 @ W2r + b2 )       [N x 256]
    gemm_wmma<true, true, false, true><<<dim3(gemmRows, HDIM / 64), 256, 0, stream>>>(
        buf2, pW2l, HDIM, buf1, pW2r, HDIM, deg, nullptr, nullptr, b2, buf0, NNODES, HDIM);

    // 6) h3 = h2 @ Wm1 + bm1  (pre-BN)                          [N x 256]
    gemm_wmma<false, false, false, false><<<dim3(gemmRows, HDIM / 64), 256, 0, stream>>>(
        buf0, pWm1, HDIM, nullptr, nullptr, 0, nullptr, nullptr, nullptr, bm1,
        buf2, NNODES, HDIM);

    // 7) batchnorm statistics -> folded scale/shift
    bn_stats<<<HDIM, 256, 0, stream>>>(buf2, gam, bet, bnscale, bnshift, NNODES, HDIM);

    // 8) out = relu(bn(h3)) @ Wm2 + bm2                         [N x 128]
    gemm_wmma<false, false, true, false><<<dim3(gemmRows, EMBD / 64), 256, 0, stream>>>(
        buf2, pWm2, HDIM, nullptr, nullptr, 0, nullptr, bnscale, bnshift, bm2,
        out, NNODES, EMBD);
}
